// MambaBlock_21620865368263
// MI455X (gfx1250) — matrix operand order repaired and checked by reading, not compiled, not run on hardware
//
#include <hip/hip_runtime.h>
#include <hip/hip_bf16.h>
#include <cmath>

// ---------------- problem constants (match reference) ----------------
#define BATCH   2
#define SEQ     2048
#define D_MODEL 768
#define D_INNER 1536
#define D_STATE 16
#define D_CONV  4
#define DT_RANK 48
#define DT_PAD  64                   // dt rank zero-padded to K=64 for clean WMMA K-loop
#define NTOK    (BATCH * SEQ)        // 4096 tokens
#define XZ_W    (2 * D_INNER)        // 3072
#define DBL_W   (DT_RANK + 2 * D_STATE)  // 80

typedef _Float16 v16h __attribute__((ext_vector_type(16)));
typedef _Float16 v8h  __attribute__((ext_vector_type(8)));
typedef float    v8f  __attribute__((ext_vector_type(8)));
typedef int      v4i  __attribute__((ext_vector_type(4)));

__device__ inline v8h zero8() { v8h z = {}; return z; }

// ---- gfx1250 async global->LDS copy (ASYNCcnt), with sync fallback ----
#if defined(__has_builtin)
#  if __has_builtin(__builtin_amdgcn_global_load_async_to_lds_b128)
#    define HAVE_ASYNC_LDS 1
#  else
#    define HAVE_ASYNC_LDS 0
#  endif
#  if __has_builtin(__builtin_amdgcn_s_wait_asynccnt)
#    define HAVE_ASYNC_WAIT 1
#  else
#    define HAVE_ASYNC_WAIT 0
#  endif
#else
#  define HAVE_ASYNC_LDS 0
#  define HAVE_ASYNC_WAIT 0
#endif

// builtin signature (from clang diagnostic): (int4 AS1* gsrc, int4 AS3* ldst, imm, imm)
typedef __attribute__((address_space(1))) v4i* g4_ptr;
typedef __attribute__((address_space(3))) v4i* l4_ptr;

__device__ inline void wait_async_lds() {
#if HAVE_ASYNC_WAIT
    __builtin_amdgcn_s_wait_asynccnt(0);
#else
    asm volatile("s_wait_asynccnt 0x0" ::: "memory");
#endif
}

// ---------------------------------------------------------------------
// small utility kernels
// ---------------------------------------------------------------------
__global__ void k_f32_to_f16(const float* __restrict__ in, _Float16* __restrict__ out, int n) {
    int i = blockIdx.x * 256 + threadIdx.x;
    if (i < n) out[i] = (_Float16)in[i];
}

// dt_proj_w [1536,48] f32 -> [1536,64] f16 with zero pad in K
__global__ void k_pad_dtw(const float* __restrict__ in, _Float16* __restrict__ out) {
    int i = blockIdx.x * 256 + threadIdx.x;
    if (i >= D_INNER * DT_PAD) return;
    int r = i / DT_PAD, c = i % DT_PAD;
    out[i] = (c < DT_RANK) ? (_Float16)in[r * DT_RANK + c] : (_Float16)0;
}

__global__ void k_fill_h0(_Float16* __restrict__ p, int n) {
    int i = blockIdx.x * 256 + threadIdx.x;
    if (i < n) p[i] = (_Float16)0;
}

// ---------------------------------------------------------------------
// LayerNorm over last dim (768), one 256-thread block per token row.
// ---------------------------------------------------------------------
__global__ __launch_bounds__(256) void k_layernorm(
    const float* __restrict__ x, const float* __restrict__ w, const float* __restrict__ b,
    _Float16* __restrict__ xn_h)
{
    __shared__ float ssum[256], ssq[256];
    const int row = blockIdx.x;
    const int tid = threadIdx.x;
    const float* xr = x + (size_t)row * D_MODEL;
    float s = 0.f, q = 0.f;
    for (int i = tid; i < D_MODEL; i += 256) { float v = xr[i]; s += v; q += v * v; }
    ssum[tid] = s; ssq[tid] = q;
    __syncthreads();
    for (int st = 128; st > 0; st >>= 1) {
        if (tid < st) { ssum[tid] += ssum[tid + st]; ssq[tid] += ssq[tid + st]; }
        __syncthreads();
    }
    const float mean = ssum[0] * (1.f / D_MODEL);
    const float var  = ssq[0] * (1.f / D_MODEL) - mean * mean;
    const float rstd = rsqrtf(var + 1e-5f);
    for (int i = tid; i < D_MODEL; i += 256)
        xn_h[(size_t)row * D_MODEL + i] = (_Float16)((xr[i] - mean) * rstd * w[i] + b[i]);
}

// ---------------------------------------------------------------------
// WMMA fragment helpers (CDNA5 §7.12.2 16-bit layouts, wave32).
// K is always a multiple of 32 here -> no predication anywhere.
// ---------------------------------------------------------------------
// A 16x32 f16: lane m in [0,16) holds row m, K {0..7,16..23}; lane m+16: {8..15,24..31}
__device__ inline v16h load_a_frag(const _Float16* __restrict__ A, int ld, int row,
                                   int kb, int hs) {
    const _Float16* p = A + (size_t)row * ld + kb + hs * 8;
    v8h c0 = *(const v8h*)(p);
    v8h c1 = *(const v8h*)(p + 16);
    return __builtin_shufflevector(c0, c1, 0,1,2,3,4,5,6,7,8,9,10,11,12,13,14,15);
}

// B 32x16 f16 from LDS tile [64 rows][32 k]: lane n holds col n, k {0..7,16..23} (lanes 0-15)
// or k {8..15,24..31} (lanes 16-31) -> halves at row*32 + hs*8 and +16.
__device__ inline v16h load_b_frag_lds(const _Float16* __restrict__ Bs, int lrow, int hs) {
    const _Float16* p = Bs + lrow * 32 + hs * 8;
    v8h c0 = *(const v8h*)(p);
    v8h c1 = *(const v8h*)(p + 16);
    return __builtin_shufflevector(c0, c1, 0,1,2,3,4,5,6,7,8,9,10,11,12,13,14,15);
}

// ---------------------------------------------------------------------
// WMMA GEMM:  C[M,N] = A[M,K](f16) * W[N,K](f16)^T  (f32 accumulate)
// Block = 256 threads = 8 waves. Each wave: TWO 16-row M tiles; block strip = 64 N cols
// (block footprint 256 x 64). B tile (64x32 = 4KB) staged in LDS per K-step, double
// buffered; staged with GLOBAL_LOAD_ASYNC_TO_LDS_B128 when available (full-N kernels).
// REQUIRES: M % 256 == 0 (true for all launches here: M = 4096).
// EPI: 0 none, 1 softplus(c+bias[n]), 2 c+resid.  STOREH: f16 side copy (cc<HCOLS, ldh).
// ---------------------------------------------------------------------
template<int N, int K, int EPI, bool STOREH, int HCOLS>
__global__ __launch_bounds__(256) void k_gemm_wmma(
    const _Float16* __restrict__ A, int lda,
    const _Float16* __restrict__ W, int ldb,
    float* __restrict__ C, _Float16* __restrict__ Ch, int ldc, int ldh,
    const float* __restrict__ bias, const float* __restrict__ resid, int ldr)
{
    __shared__ _Float16 Bs[2][64 * 32];          // 2 x 4KB ping-pong

    const int tid   = threadIdx.x;
    const int lane  = tid & 31;
    const int wave  = tid >> 5;
    const int hs    = lane >> 4;                 // half-wave select
    const int col   = lane & 15;
    const int mbase = (blockIdx.x * 8 + wave) * 32;   // two tiles: mbase, mbase+16
    const int n0    = blockIdx.y * 64;
    constexpr bool FULLN = (N % 64 == 0);        // all 4 n-tiles always valid
    constexpr bool ASYNC = FULLN && (HAVE_ASYNC_LDS != 0);

    // staging coords: 64 rows x 4 chunks of 8 halves; one b128 per thread
    const int srow  = tid >> 2;
    const int schnk = (tid & 3) * 8;
    const bool svalid = FULLN || (n0 + srow < N);
    const _Float16* sptr = W + (size_t)(n0 + srow) * ldb + schnk;
    _Float16* const sdst0 = &Bs[0][srow * 32 + schnk];
    _Float16* const sdst1 = &Bs[1][srow * 32 + schnk];

    v8f acc[2][4] = {};

    // prime buffer 0 with K-step 0
    if constexpr (ASYNC) {
        __builtin_amdgcn_global_load_async_to_lds_b128((g4_ptr)sptr, (l4_ptr)sdst0, 0, 0);
    } else {
        v8h v = svalid ? *(const v8h*)(sptr) : zero8();
        *(v8h*)(sdst0) = v;
    }

    int buf = 0;
    for (int kb = 0; kb < K; kb += 32, buf ^= 1) {
        if constexpr (ASYNC) wait_async_lds();
        __syncthreads();                          // staged tile visible to all waves
        if (kb + 32 < K) {                        // prefetch next K tile -> other buffer
            _Float16* sdst = (buf == 0) ? sdst1 : sdst0;
            if constexpr (ASYNC) {
                __builtin_amdgcn_global_load_async_to_lds_b128(
                    (g4_ptr)(sptr + kb + 32), (l4_ptr)sdst, 0, 0);
            } else {
                v8h v = svalid ? *(const v8h*)(sptr + kb + 32) : zero8();
                *(v8h*)(sdst) = v;
            }
        }

        const v16h a0 = load_a_frag(A, lda, mbase +      col, kb, hs);
        const v16h a1 = load_a_frag(A, lda, mbase + 16 + col, kb, hs);
        const _Float16* bsp = &Bs[buf][0];

        if constexpr (FULLN) {
            // load all B fragments first -> single dscnt wait, 8 back-to-back WMMAs
            const v16h b0 = load_b_frag_lds(bsp,  0 + col, hs);
            const v16h b1 = load_b_frag_lds(bsp, 16 + col, hs);
            const v16h b2 = load_b_frag_lds(bsp, 32 + col, hs);
            const v16h b3 = load_b_frag_lds(bsp, 48 + col, hs);
            acc[0][0] = __builtin_amdgcn_wmma_f32_16x16x32_f16(false, a0, false, b0, (short)0, acc[0][0], false, false);
            acc[0][1] = __builtin_amdgcn_wmma_f32_16x16x32_f16(false, a0, false, b1, (short)0, acc[0][1], false, false);
            acc[0][2] = __builtin_amdgcn_wmma_f32_16x16x32_f16(false, a0, false, b2, (short)0, acc[0][2], false, false);
            acc[0][3] = __builtin_amdgcn_wmma_f32_16x16x32_f16(false, a0, false, b3, (short)0, acc[0][3], false, false);
            acc[1][0] = __builtin_amdgcn_wmma_f32_16x16x32_f16(false, a1, false, b0, (short)0, acc[1][0], false, false);
            acc[1][1] = __builtin_amdgcn_wmma_f32_16x16x32_f16(false, a1, false, b1, (short)0, acc[1][1], false, false);
            acc[1][2] = __builtin_amdgcn_wmma_f32_16x16x32_f16(false, a1, false, b2, (short)0, acc[1][2], false, false);
            acc[1][3] = __builtin_amdgcn_wmma_f32_16x16x32_f16(false, a1, false, b3, (short)0, acc[1][3], false, false);
        } else {
            #pragma unroll
            for (int j = 0; j < 4; ++j) {
                if (n0 + j * 16 < N) {
                    const v16h b = load_b_frag_lds(bsp, j * 16 + col, hs);
                    acc[0][j] = __builtin_amdgcn_wmma_f32_16x16x32_f16(false, a0, false, b, (short)0, acc[0][j], false, false);
                    acc[1][j] = __builtin_amdgcn_wmma_f32_16x16x32_f16(false, a1, false, b, (short)0, acc[1][j], false, false);
                }
            }
        }
    }

    #pragma unroll
    for (int i = 0; i < 2; ++i) {
        const int m0 = mbase + i * 16;
        #pragma unroll
        for (int j = 0; j < 4; ++j) {
            const int n = n0 + j * 16;
            if (!FULLN && n >= N) continue;
            const int cc = n + col;
            #pragma unroll
            for (int r = 0; r < 8; ++r) {
                const int rr = m0 + r + hs * 8;   // C/D layout: VGPR r -> row r (+8 upper)
                float v = acc[i][j][r];
                if (EPI == 1) {                    // softplus(c + bias)
                    v += bias[cc];
                    v = (v > 20.f) ? v : log1pf(__expf(v));
                }
                if (EPI == 2) v += resid[(size_t)rr * ldr + cc];
                C[(size_t)rr * ldc + cc] = v;
                if (STOREH && cc < HCOLS) Ch[(size_t)rr * ldh + cc] = (_Float16)v;
            }
        }
    }
}

// ---------------------------------------------------------------------
// Causal depthwise conv1d (taps=4) + SiLU.  One thread per (b,t,d).
// ---------------------------------------------------------------------
__global__ __launch_bounds__(256) void k_conv_silu(
    const float* __restrict__ xz, const float* __restrict__ cw, const float* __restrict__ cb,
    float* __restrict__ uc, _Float16* __restrict__ uc_h)
{
    const long idx = (long)blockIdx.x * 256 + threadIdx.x;
    if (idx >= (long)NTOK * D_INNER) return;
    const int d   = (int)(idx % D_INNER);
    const int tok = (int)(idx / D_INNER);
    const int t   = tok % SEQ;
    float acc = cb[d];
    #pragma unroll
    for (int k = 0; k < D_CONV; ++k) {
        const int tt = t + k - (D_CONV - 1);
        if (tt >= 0) acc += cw[d * D_CONV + k] * xz[((size_t)(tok + k - (D_CONV - 1))) * XZ_W + d];
    }
    acc = acc / (1.f + __expf(-acc));   // SiLU
    uc[idx]   = acc;
    uc_h[idx] = (_Float16)acc;
}

// ---------------------------------------------------------------------
// Selective scan: one (b,d) channel per 16-lane half-wave, lane = state s.
// h_s[t] = exp(delta*A_s) h_s[t-1] + delta*u*B_t[s];  y = sum_s h_s*C_t[s]
// ---------------------------------------------------------------------
__global__ __launch_bounds__(256) void k_scan(
    const float* __restrict__ delta,   // [NTOK, D_INNER]
    const float* __restrict__ uc,      // [NTOK, D_INNER]
    const float* __restrict__ dbl,     // [NTOK, DBL_W]  (B at 48, C at 64)
    const float* __restrict__ xz,      // z at col D_INNER, stride XZ_W
    const float* __restrict__ A_log,   // [D_INNER, D_STATE]
    const float* __restrict__ Dvec,    // [D_INNER]
    _Float16* __restrict__ y_h)        // [NTOK, D_INNER]
{
    const int gid = blockIdx.x * 256 + threadIdx.x;
    const int ch  = gid >> 4;                 // (b,d) channel, 0..3071
    const int s   = gid & 15;                 // state index
    if (ch >= BATCH * D_INNER) return;
    const int b = ch / D_INNER;
    const int d = ch % D_INNER;

    const float Aneg = -__expf(A_log[d * D_STATE + s]);
    const float Dd   = Dvec[d];

    const float* dP = delta + (size_t)b * SEQ * D_INNER + d;
    const float* uP = uc    + (size_t)b * SEQ * D_INNER + d;
    const float* bP = dbl   + (size_t)b * SEQ * DBL_W + DT_RANK + s;
    const float* cP = bP + D_STATE;
    const float* zP = xz    + (size_t)b * SEQ * XZ_W + D_INNER + d;
    _Float16*    yP = y_h   + (size_t)b * SEQ * D_INNER + d;

    float h = 0.f;
    for (int t = 0; t < SEQ; ++t) {
        if (t + 16 < SEQ) {                         // stay ahead of the recurrence
            __builtin_prefetch(dP + 16 * D_INNER, 0, 1);
            __builtin_prefetch(bP + 16 * DBL_W,  0, 1);
        }
        const float dt = *dP;
        const float ut = *uP;
        const float Bt = *bP;
        const float Ct = *cP;
        const float dA = __expf(dt * Aneg);
        h = fmaf(dA, h, (dt * ut) * Bt);
        float ys = h * Ct;
        ys += __shfl_xor(ys, 8, 32);               // 16-lane tree (xor stays in group)
        ys += __shfl_xor(ys, 4, 32);
        ys += __shfl_xor(ys, 2, 32);
        ys += __shfl_xor(ys, 1, 32);
        if (s == 0) {
            const float z = *zP;
            float y = ys + Dd * ut;
            y *= z / (1.f + __expf(-z));           // * silu(z)
            *yP = (_Float16)y;
        }
        dP += D_INNER; uP += D_INNER; bP += DBL_W; cP += DBL_W;
        zP += XZ_W;    yP += D_INNER;
    }
}

// ---------------------------------------------------------------------
extern "C" void kernel_launch(void* const* d_in, const int* in_sizes, int n_in,
                              void* d_out, int out_size, void* d_ws, size_t ws_size,
                              hipStream_t stream) {
    (void)in_sizes; (void)n_in; (void)out_size; (void)ws_size;
    const float* x        = (const float*)d_in[0];
    const float* ln_w     = (const float*)d_in[1];
    const float* ln_b     = (const float*)d_in[2];
    const float* in_w     = (const float*)d_in[3];   // [3072,768]
    const float* conv_w   = (const float*)d_in[4];   // [1536,4]
    const float* conv_b   = (const float*)d_in[5];
    const float* xp_w     = (const float*)d_in[6];   // [80,1536]
    const float* dt_w     = (const float*)d_in[7];   // [1536,48]
    const float* dt_b     = (const float*)d_in[8];
    const float* A_log    = (const float*)d_in[9];   // [1536,16]
    const float* Dvec     = (const float*)d_in[10];
    const float* out_w    = (const float*)d_in[11];  // [768,1536]
    float* out            = (float*)d_out;

    // ---- workspace layout (256B aligned) ----
    char* ws = (char*)d_ws;
    size_t off = 0;
    auto alloc = [&](size_t bytes) { size_t r = off; off += (bytes + 255) & ~(size_t)255; return r; };
    _Float16* w_in_h  = (_Float16*)(ws + alloc((size_t)XZ_W * D_MODEL * 2));
    _Float16* w_xp_h  = (_Float16*)(ws + alloc((size_t)DBL_W * D_INNER * 2));
    _Float16* w_dt_h  = (_Float16*)(ws + alloc((size_t)D_INNER * DT_PAD * 2));   // padded K
    _Float16* w_out_h = (_Float16*)(ws + alloc((size_t)D_MODEL * D_INNER * 2));
    _Float16* xn_h    = (_Float16*)(ws + alloc((size_t)NTOK * D_MODEL * 2));
    float*    xz      = (float*)   (ws + alloc((size_t)NTOK * XZ_W * 4));
    float*    uc      = (float*)   (ws + alloc((size_t)NTOK * D_INNER * 4));
    _Float16* uc_h    = (_Float16*)(ws + alloc((size_t)NTOK * D_INNER * 2));
    float*    dbl     = (float*)   (ws + alloc((size_t)NTOK * DBL_W * 4));
    _Float16* dt_h    = (_Float16*)(ws + alloc((size_t)NTOK * DT_PAD * 2));      // padded dt
    float*    delta   = (float*)   (ws + alloc((size_t)NTOK * D_INNER * 4));
    _Float16* y_h     = (_Float16*)(ws + alloc((size_t)NTOK * D_INNER * 2));

    auto cvt = [&](const float* src, _Float16* dst, int n) {
        k_f32_to_f16<<<(n + 255) / 256, 256, 0, stream>>>(src, dst, n);
    };
    cvt(in_w,  w_in_h,  XZ_W * D_MODEL);
    cvt(xp_w,  w_xp_h,  DBL_W * D_INNER);
    cvt(out_w, w_out_h, D_MODEL * D_INNER);
    k_pad_dtw<<<(D_INNER * DT_PAD + 255) / 256, 256, 0, stream>>>(dt_w, w_dt_h);
    k_fill_h0<<<(NTOK * DT_PAD + 255) / 256, 256, 0, stream>>>(dt_h, NTOK * DT_PAD);

    // 1) layernorm -> xn (f16)
    k_layernorm<<<NTOK, 256, 0, stream>>>(x, ln_w, ln_b, xn_h);

    // 2) in_proj: xz[4096,3072] = xn * in_w^T
    k_gemm_wmma<XZ_W, D_MODEL, 0, false, 0>
        <<<dim3(NTOK / 256, XZ_W / 64), 256, 0, stream>>>(
        xn_h, D_MODEL, w_in_h, D_MODEL, xz, nullptr, XZ_W, 0,
        nullptr, nullptr, 0);

    // 3) causal depthwise conv + SiLU -> uc (f32 + f16)
    {
        long n = (long)NTOK * D_INNER;
        k_conv_silu<<<(unsigned)((n + 255) / 256), 256, 0, stream>>>(xz, conv_w, conv_b, uc, uc_h);
    }

    // 4) x_proj: dbl[4096,80] = uc * xp_w^T ; dt cols (<48) side-stored f16 into padded dt_h
    k_gemm_wmma<DBL_W, D_INNER, 0, true, DT_RANK>
        <<<dim3(NTOK / 256, (DBL_W + 63) / 64), 256, 0, stream>>>(
        uc_h, D_INNER, w_xp_h, D_INNER, dbl, dt_h, DBL_W, DT_PAD,
        nullptr, nullptr, 0);

    // 5) dt_proj + softplus: delta[4096,1536] = softplus(dt * dt_w^T + dt_b), K padded to 64
    k_gemm_wmma<D_INNER, DT_PAD, 1, false, 0>
        <<<dim3(NTOK / 256, D_INNER / 64), 256, 0, stream>>>(
        dt_h, DT_PAD, w_dt_h, DT_PAD, delta, nullptr, D_INNER, 0,
        dt_b, nullptr, 0);

    // 6) selective scan + D-skip + SiLU(z) gating -> y (f16)
    k_scan<<<(BATCH * D_INNER * D_STATE) / 256, 256, 0, stream>>>(
        delta, uc, dbl, xz, A_log, Dvec, y_h);

    // 7) out_proj + residual: out[4096,768] = y * out_w^T + x
    k_gemm_wmma<D_MODEL, D_INNER, 2, false, 0>
        <<<dim3(NTOK / 256, D_MODEL / 64), 256, 0, stream>>>(
        y_h, D_INNER, w_out_h, D_INNER, out, nullptr, D_MODEL, 0,
        nullptr, x, D_MODEL);
}
